// Net_35364760715550
// MI455X (gfx1250) — matrix operand order repaired and checked by reading, not compiled, not run on hardware
//
#include <hip/hip_runtime.h>
#include <hip/hip_bf16.h>

#define NNODES 100000
#define NEDGES 1600000
#define FIN    64
#define HID    256
#define NMONO  4000
#define NPOLY  1000
#define ATT_SCALE 0.0625f   // 256^-0.5

typedef __attribute__((ext_vector_type(16))) __bf16 v16bf;
typedef __attribute__((ext_vector_type(8)))  __bf16 v8bf;
typedef __attribute__((ext_vector_type(8)))  float  v8f;

// ======================= elementwise copy =======================
__global__ void copy_f32v4(const float4* __restrict__ in, float4* __restrict__ out, long n4) {
  long i = blockIdx.x * (long)blockDim.x + threadIdx.x;
  if (i < n4) out[i] = in[i];
}

// ======================= edge scatter-add =======================
__global__ void edge_agg(const float* __restrict__ x, const int* __restrict__ src,
                         const int* __restrict__ dst, float* __restrict__ acc, int H) {
  long tid = blockIdx.x * (long)blockDim.x + threadIdx.x;
  int hf = H >> 2;
  long total = (long)NEDGES * hf;
  if (tid >= total) return;
  int e = (int)(tid / hf);
  int c = (int)(tid - (long)e * hf);
  int s = src[e], d = dst[e];
  float4 v = *(const float4*)(x + (long)s * H + (c << 2));
  float* p = acc + (long)d * H + (c << 2);
  atomicAdd(p + 0, v.x);
  atomicAdd(p + 1, v.y);
  atomicAdd(p + 2, v.z);
  atomicAdd(p + 3, v.w);
}

// ======================= weight packing =======================
// Pack f32 W[K x Ncols] into wave32 WMMA bf16 B-matrix tiles (see round-0 comment).
__global__ void pack_b(const float* __restrict__ W, unsigned int* __restrict__ P,
                       int K, int Ncols) {
  int KT = K >> 5, NT = Ncols >> 4;
  int t = blockIdx.x * blockDim.x + threadIdx.x;
  int total = KT * NT * 32;
  if (t >= total) return;
  int lane = t & 31;
  int tile = t >> 5;
  int nt = tile % NT, kt = tile / NT;
  int col = (nt << 4) + (lane & 15);
  int kb  = (kt << 5) + ((lane & 16) ? 8 : 0);
  unsigned int* dstp = P + (size_t)tile * 256 + lane * 8;
#pragma unroll
  for (int j = 0; j < 8; ++j) {
    union { __bf16 h; unsigned short u; } a, b;
    a.h = (__bf16)W[(long)(kb + 2 * j + (j >= 4 ? 8 : 0)) * Ncols + col];
    b.h = (__bf16)W[(long)(kb + 2 * j + 1 + (j >= 4 ? 8 : 0)) * Ncols + col];
    dstp[j] = (unsigned)a.u | ((unsigned)b.u << 16);
  }
}

// ======================= WMMA GEMM =======================
__device__ __forceinline__ v16bf cvt_a_f32(const float* __restrict__ p) {
  float4 f0 = *(const float4*)(p);
  float4 f1 = *(const float4*)(p + 4);
  float4 f2 = *(const float4*)(p + 16);
  float4 f3 = *(const float4*)(p + 20);
  v16bf v;
  v[0] = (__bf16)f0.x;  v[1] = (__bf16)f0.y;  v[2] = (__bf16)f0.z;  v[3] = (__bf16)f0.w;
  v[4] = (__bf16)f1.x;  v[5] = (__bf16)f1.y;  v[6] = (__bf16)f1.z;  v[7] = (__bf16)f1.w;
  v[8] = (__bf16)f2.x;  v[9] = (__bf16)f2.y;  v[10] = (__bf16)f2.z; v[11] = (__bf16)f2.w;
  v[12] = (__bf16)f3.x; v[13] = (__bf16)f3.y; v[14] = (__bf16)f3.z; v[15] = (__bf16)f3.w;
  return v;
}

__device__ __forceinline__ v16bf load_a_bf16(const __bf16* __restrict__ p) {
  union { v8bf h[2]; v16bf v; } u;
  u.h[0] = *(const v8bf*)(p);
  u.h[1] = *(const v8bf*)(p + 16);
  return u.v;
}

__device__ __forceinline__ v16bf load_b2(const uint4* __restrict__ p) {
  union { uint4 u[2]; v16bf v; } w;
  w.u[0] = p[0];
  w.u[1] = p[1];
  return w.v;
}

// C[M x Ncols] = act( A[M x K] @ W + bias ), optional per-row scale.
// Block = 8 waves sharing one 64-col strip; the strip's B tiles (<=32KB) are
// staged once into LDS with global_load_async_to_lds_b128 (ASYNCcnt), then each
// wave runs a 16x64 WMMA strip reading B via ds_load.
template <bool A_IS_BF16>
__global__ void gemm_wmma(const void* __restrict__ Aptr, const uint4* __restrict__ Bp,
                          const float* __restrict__ bias, const float* __restrict__ rowscale,
                          float* __restrict__ outf, __bf16* __restrict__ outh,
                          int Mrows, int K, int Ncols, int relu) {
  __shared__ uint4 smem[2048];                       // 32 KB: up to KT=8 x 4 tiles x 1KB
  int lane = threadIdx.x & 31;
  int waveId = threadIdx.x >> 5;                     // 0..7
  int strips = Ncols >> 6;
  int rowTiles = (Mrows + 15) >> 4;
  int strip = blockIdx.x % strips;
  int rt = (blockIdx.x / strips) * 8 + waveId;
  int NT = Ncols >> 4, KT = K >> 5;

  // ---- cooperative async stage of this strip's B tiles into LDS ----
  unsigned lbase = (unsigned)(size_t)(void*)smem;    // LDS byte offset (flat low bits)
  int totalChunks = KT * 4 * 2;                      // 512B per async b128 wave-op
  for (int ch = waveId; ch < totalChunks; ch += 8) {
    int ltile = ch >> 1;
    int half = ch & 1;
    int kt = ltile >> 2, c = ltile & 3;
    int gtile = kt * NT + (strip << 2) + c;
    const char* gp = (const char*)(Bp + (size_t)gtile * 64) + half * 512 + lane * 16;
    unsigned laddr = lbase + ltile * 1024 + half * 512 + lane * 16;
    asm volatile("global_load_async_to_lds_b128 %0, %1, off"
                 :: "v"(laddr), "v"(gp) : "memory");
  }
  asm volatile("s_wait_asynccnt 0x0" ::: "memory");
  __syncthreads();

  // ---- per-wave 16x64 WMMA strip ----
  bool active = (rt < rowTiles);
  int rtc = active ? rt : (rowTiles - 1);
  int row0 = rtc << 4;
  int arow = row0 + (lane & 15);
  if (arow >= Mrows) arow = Mrows - 1;               // clamp partial-tile loads
  int koff = (lane & 16) ? 8 : 0;
  int esz = A_IS_BF16 ? 2 : 4;
  const char* aPtr = (const char*)Aptr + ((long)arow * K + koff) * esz;
  const uint4* bPtr = smem + lane * 2;               // ds loads; +64 per tile, +256 per kt

  v8f acc0 = {}, acc1 = {}, acc2 = {}, acc3 = {};
  for (int kt = 0; kt < KT; ++kt) {
    v16bf a = A_IS_BF16 ? load_a_bf16((const __bf16*)aPtr) : cvt_a_f32((const float*)aPtr);
    if (kt + 1 < KT) __builtin_prefetch(aPtr + 32 * esz, 0, 3);
    v16bf b0 = load_b2(bPtr + 0);
    v16bf b1 = load_b2(bPtr + 64);
    v16bf b2 = load_b2(bPtr + 128);
    v16bf b3 = load_b2(bPtr + 192);
    acc0 = __builtin_amdgcn_wmma_f32_16x16x32_bf16(false, a, false, b0, (short)0, acc0, false, false);
    acc1 = __builtin_amdgcn_wmma_f32_16x16x32_bf16(false, a, false, b1, (short)0, acc1, false, false);
    acc2 = __builtin_amdgcn_wmma_f32_16x16x32_bf16(false, a, false, b2, (short)0, acc2, false, false);
    acc3 = __builtin_amdgcn_wmma_f32_16x16x32_bf16(false, a, false, b3, (short)0, acc3, false, false);
    aPtr += 32 * esz;
    bPtr += 256;
  }
  if (!active) return;                               // wave-uniform, after all barriers

  int crow0 = row0 + ((lane & 16) ? 8 : 0);
  int colb = (strip << 6) + (lane & 15);
  v8f accs[4] = {acc0, acc1, acc2, acc3};
#pragma unroll
  for (int c = 0; c < 4; ++c) {
    int col = colb + (c << 4);
    float bv = bias ? bias[col] : 0.0f;
#pragma unroll
    for (int r = 0; r < 8; ++r) {
      int row = crow0 + r;
      if (row >= Mrows) continue;
      float val = accs[c][r] + bv;
      if (rowscale) val *= rowscale[row];
      if (relu) val = fmaxf(val, 0.0f);
      long idx = (long)row * Ncols + col;
      if (outf) outf[idx] = val;
      if (outh) outh[idx] = (__bf16)val;
    }
  }
}

// ======================= monomer pooling =======================
__global__ void monomer_pool(const float* __restrict__ x, const float* __restrict__ ratio,
                             const int* __restrict__ batch, float* __restrict__ memb,
                             __bf16* __restrict__ membh, float* __restrict__ frac,
                             int* __restrict__ bidx) {
  int m = blockIdx.x, h = threadIdx.x;
  long base = (long)m * 25;
  float s = 0.0f;
  for (int i = 0; i < 25; ++i) s += x[(base + i) * HID + h];
  s *= (1.0f / 25.0f);
  memb[(long)m * HID + h] = s;
  membh[(long)m * HID + h] = (__bf16)s;
  if (h == 0) {
    float mn = ratio[base];
    for (int i = 1; i < 25; ++i) mn = fminf(mn, ratio[base + i]);
    frac[m] = mn;
    bidx[m] = batch[base];
  }
}

// ======================= per-polymer k-sum =======================
__global__ void ksum_kernel(const float* __restrict__ k, float* __restrict__ ks) {
  int b = blockIdx.x, h = threadIdx.x;
  float s = 0.0f;
#pragma unroll
  for (int i = 0; i < 4; ++i) s += k[(long)(4 * b + i) * HID + h];
  ks[(long)b * HID + h] = s;
}

// ======================= attention softmax + weighted v =======================
__global__ void attn_pool(const float* __restrict__ q, const float* __restrict__ v,
                          const float* __restrict__ ks, float* __restrict__ poly) {
  int b = blockIdx.x, h = threadIdx.x;
  float kk = ks[(long)b * HID + h] * ATT_SCALE;
  float e[4], emax = -3.4e38f;
#pragma unroll
  for (int i = 0; i < 4; ++i) {
    e[i] = q[(long)(4 * b + i) * HID + h] * kk;
    emax = fmaxf(emax, e[i]);
  }
  float denom = 0.0f;
#pragma unroll
  for (int i = 0; i < 4; ++i) { e[i] = __expf(e[i] - emax); denom += e[i]; }
  float inv = 1.0f / (denom + 1e-16f);
  float s = 0.0f;
#pragma unroll
  for (int i = 0; i < 4; ++i) s += v[(long)(4 * b + i) * HID + h] * e[i] * inv;
  poly[(long)b * HID + h] = s;
}

// ======================= final projection to scalar =======================
__global__ void head_out(const __bf16* __restrict__ hh, const float* __restrict__ Wo2,
                         const float* __restrict__ bo2, float* __restrict__ out) {
  int b = blockIdx.x * 8 + (threadIdx.x >> 5);
  int lane = threadIdx.x & 31;
  if (b >= NPOLY) return;
  float s = 0.0f;
  for (int j = lane; j < HID; j += 32) s += (float)hh[(long)b * HID + j] * Wo2[j];
#pragma unroll
  for (int off = 16; off > 0; off >>= 1) s += __shfl_down(s, off, 32);
  if (lane == 0) out[b] = s + bo2[0];
}

// ======================= host orchestration =======================
extern "C" void kernel_launch(void* const* d_in, const int* in_sizes, int n_in,
                              void* d_out, int out_size, void* d_ws, size_t ws_size,
                              hipStream_t stream) {
  const float* x_in  = (const float*)d_in[0];
  const float* ratio = (const float*)d_in[1];
  const int*   eidx  = (const int*)d_in[2];
  const int*   batch = (const int*)d_in[4];
  const float* Wg0a = (const float*)d_in[5],  *bg0a = (const float*)d_in[6];
  const float* Wg0b = (const float*)d_in[7],  *bg0b = (const float*)d_in[8];
  const float* Wg1a = (const float*)d_in[9],  *bg1a = (const float*)d_in[10];
  const float* Wg1b = (const float*)d_in[11], *bg1b = (const float*)d_in[12];
  const float* Wg2a = (const float*)d_in[13], *bg2a = (const float*)d_in[14];
  const float* Wg2b = (const float*)d_in[15], *bg2b = (const float*)d_in[16];
  const float* Wq = (const float*)d_in[17], *bq = (const float*)d_in[18];
  const float* Wk = (const float*)d_in[19], *bk = (const float*)d_in[20];
  const float* Wv = (const float*)d_in[21], *bv = (const float*)d_in[22];
  const float* Wo1 = (const float*)d_in[23], *bo1 = (const float*)d_in[24];
  const float* Wo2 = (const float*)d_in[25], *bo2 = (const float*)d_in[26];
  const int* src = eidx;
  const int* dst = eidx + NEDGES;

  // ---- workspace carve-up ----
  char* base = (char*)d_ws;
  size_t off = 0;
  auto carve = [&](size_t bytes) -> char* {
    char* p = base + off;
    off += (bytes + 255) & ~(size_t)255;
    return p;
  };
  float*  xbuf  = (float*) carve((size_t)NNODES * HID * 4);
  float*  hbuf  = (float*) carve((size_t)NNODES * HID * 4);
  __bf16* h1buf = (__bf16*)carve((size_t)NNODES * HID * 2);
  auto packed_bytes = [](int K, int Ncols) { return (size_t)(K / 32) * (Ncols / 16) * 1024; };
  uint4* pg0a = (uint4*)carve(packed_bytes(FIN, HID));
  uint4* pg0b = (uint4*)carve(packed_bytes(HID, HID));
  uint4* pg1a = (uint4*)carve(packed_bytes(HID, HID));
  uint4* pg1b = (uint4*)carve(packed_bytes(HID, HID));
  uint4* pg2a = (uint4*)carve(packed_bytes(HID, HID));
  uint4* pg2b = (uint4*)carve(packed_bytes(HID, HID));
  uint4* pq   = (uint4*)carve(packed_bytes(HID, HID));
  uint4* pk   = (uint4*)carve(packed_bytes(HID, HID));
  uint4* pv   = (uint4*)carve(packed_bytes(HID, HID));
  uint4* po1  = (uint4*)carve(packed_bytes(HID, HID));
  float*  memb  = (float*) carve((size_t)NMONO * HID * 4);
  __bf16* membh = (__bf16*)carve((size_t)NMONO * HID * 2);
  float*  frac  = (float*) carve((size_t)NMONO * 4);
  int*    bidx  = (int*)   carve((size_t)NMONO * 4);
  float*  qb = (float*)carve((size_t)NMONO * HID * 4);
  float*  kb = (float*)carve((size_t)NMONO * HID * 4);
  float*  vb = (float*)carve((size_t)NMONO * HID * 4);
  float*  ksb  = (float*)carve((size_t)NPOLY * HID * 4);
  float*  poly = (float*)carve((size_t)NPOLY * HID * 4);
  __bf16* hh   = (__bf16*)carve((size_t)NPOLY * HID * 2);

  // ---- pack all weights into WMMA B layout ----
  auto pack = [&](const float* W, uint4* P, int K, int Ncols) {
    int threads = (K / 32) * (Ncols / 16) * 32;
    pack_b<<<(threads + 255) / 256, 256, 0, stream>>>(W, (unsigned int*)P, K, Ncols);
  };
  pack(Wg0a, pg0a, FIN, HID);
  pack(Wg0b, pg0b, HID, HID);
  pack(Wg1a, pg1a, HID, HID);
  pack(Wg1b, pg1b, HID, HID);
  pack(Wg2a, pg2a, HID, HID);
  pack(Wg2b, pg2b, HID, HID);
  pack(Wq, pq, HID, HID);
  pack(Wk, pk, HID, HID);
  pack(Wv, pv, HID, HID);
  pack(Wo1, po1, HID, HID);

  auto gemm = [&](const void* A, bool abf16, const uint4* Bp, const float* bias,
                  const float* rs, float* outf, __bf16* outh, int Mrows, int K,
                  int Ncols, int relu) {
    int rowTiles = (Mrows + 15) / 16;
    int strips = Ncols / 64;
    int blocks = ((rowTiles + 7) / 8) * strips;      // 8 waves/block share one strip
    if (abf16)
      gemm_wmma<true><<<blocks, 256, 0, stream>>>(A, Bp, bias, rs, outf, outh, Mrows, K, Ncols, relu);
    else
      gemm_wmma<false><<<blocks, 256, 0, stream>>>(A, Bp, bias, rs, outf, outh, Mrows, K, Ncols, relu);
  };
  auto agg = [&](const float* xf, float* accf, int H) {
    long total = (long)NEDGES * (H >> 2);
    edge_agg<<<(int)((total + 255) / 256), 256, 0, stream>>>(xf, src, dst, accf, H);
  };
  auto copyv = [&](const float* in, float* out, long n) {
    long n4 = n >> 2;
    copy_f32v4<<<(int)((n4 + 255) / 256), 256, 0, stream>>>((const float4*)in, (float4*)out, n4);
  };

  // ---- layer 0 (FIN=64 features) ----
  copyv(x_in, hbuf, (long)NNODES * FIN);                 // h = x
  agg(x_in, hbuf, FIN);                                  // h += sum_j x_j
  gemm(hbuf, false, pg0a, bg0a, nullptr, nullptr, h1buf, NNODES, FIN, HID, 1);
  gemm(h1buf, true, pg0b, bg0b, nullptr, xbuf, nullptr, NNODES, HID, HID, 1);

  // ---- layers 1,2 (HID features) ----
  copyv(xbuf, hbuf, (long)NNODES * HID);
  agg(xbuf, hbuf, HID);
  gemm(hbuf, false, pg1a, bg1a, nullptr, nullptr, h1buf, NNODES, HID, HID, 1);
  gemm(h1buf, true, pg1b, bg1b, nullptr, xbuf, nullptr, NNODES, HID, HID, 1);

  copyv(xbuf, hbuf, (long)NNODES * HID);
  agg(xbuf, hbuf, HID);
  gemm(hbuf, false, pg2a, bg2a, nullptr, nullptr, h1buf, NNODES, HID, HID, 1);
  gemm(h1buf, true, pg2b, bg2b, nullptr, xbuf, nullptr, NNODES, HID, HID, 1);

  // ---- monomer pooling ----
  monomer_pool<<<NMONO, HID, 0, stream>>>(xbuf, ratio, batch, memb, membh, frac, bidx);

  // ---- q/k/v (q,k fused with monomer-fraction scale) ----
  gemm(membh, true, pq, bq, frac, qb, nullptr, NMONO, HID, HID, 0);
  gemm(membh, true, pk, bk, frac, kb, nullptr, NMONO, HID, HID, 0);
  gemm(membh, true, pv, bv, nullptr, vb, nullptr, NMONO, HID, HID, 0);

  // ---- attention pooling per polymer ----
  ksum_kernel<<<NPOLY, HID, 0, stream>>>(kb, ksb);
  attn_pool<<<NPOLY, HID, 0, stream>>>(qb, vb, ksb, poly);

  // ---- output head ----
  gemm(poly, false, po1, bo1, nullptr, nullptr, hh, NPOLY, HID, HID, 1);
  head_out<<<(NPOLY + 7) / 8, 256, 0, stream>>>(hh, Wo2, bo2, (float*)d_out);
}
